// MultiHeadAttention_53326313947659
// MI455X (gfx1250) — compile-verified, matrix operand order checked by reading, not run on hardware
//
#include <hip/hip_runtime.h>
#include <hip/hip_bf16.h>

#define NUM_HEADS 12
#define HEAD_SIZE 64
#define N_EMBED   768
#define SEQ_LEN   2048
#define BATCH     4
#define LOG2E     1.4426950408889634f

typedef __attribute__((ext_vector_type(16))) __bf16 v16bf;
typedef __attribute__((ext_vector_type(8)))  float  v8f;
typedef __attribute__((ext_vector_type(4)))  unsigned int u32x4;
typedef __attribute__((ext_vector_type(8)))  int i32x8;
typedef __attribute__((ext_vector_type(4)))  int i32x4;
typedef __attribute__((ext_vector_type(4)))  unsigned int v4u;

// ---------- scalar helpers ----------
__device__ __forceinline__ unsigned short f2bf(float x) {
    unsigned int u = __float_as_uint(x);
    unsigned int r = u + 0x7FFFu + ((u >> 16) & 1u);   // round-to-nearest-even
    return (unsigned short)(r >> 16);
}

// Load a 16x32 bf16 A/B fragment from a row-major matrix (ld in elements).
// row = row0 + lane%16, half = lane/16;
// elements 0..7  <- K = k0 + 8*half + 0..7      (contiguous 16B)
// elements 8..15 <- K = k0 + 8*half + 16..23    (contiguous 16B)
__device__ __forceinline__ v16bf ldfrag(const unsigned short* base, int ld,
                                        int row0, int k0, int lane) {
    const unsigned short* p =
        base + (size_t)(row0 + (lane & 15)) * ld + k0 + ((lane >> 4) << 3);
    union { v16bf v; v4u u[2]; } un;
    un.u[0] = *(const v4u*)(p);
    un.u[1] = *(const v4u*)(p + 16);
    return un.v;
}

__device__ __forceinline__ v8f wmma_bf16(v16bf a, v16bf b, v8f c) {
    return __builtin_amdgcn_wmma_f32_16x16x32_bf16(
        false, a, false, b, (short)0, c, false, false);
}

// ---------- Tensor Data Mover: 2D bf16 tile Global -> LDS ----------
#if __has_builtin(__builtin_amdgcn_tensor_load_to_lds)
#define HAVE_TDM 1
#endif

__device__ __forceinline__ void tdm_load_tile(
    unsigned short* lds_dst, const unsigned short* gsrc,
    unsigned tile_w, unsigned tile_h, unsigned long long row_stride_elems)
{
#ifdef HAVE_TDM
    unsigned long long ga = (unsigned long long)(const void*)gsrc;
    unsigned lds_off = (unsigned)(unsigned long long)(const void*)lds_dst; // low 32b = LDS byte offset
    u32x4 g0;
    g0[0] = 1u;                                     // count=1, user descriptor
    g0[1] = lds_off;                                // lds_addr (bytes)
    g0[2] = (unsigned)(ga & 0xFFFFFFFFu);           // global_addr[31:0]
    g0[3] = (unsigned)((ga >> 32) & 0x01FFFFFFu)    // global_addr[56:32]
          | (2u << 30);                             // type = 2 ("image")
    unsigned td0 = tile_w, td1 = tile_h;            // tensor dims == tile dims (in-bounds)
    i32x8 g1;
    g1[0] = (int)(1u << 16);                        // data_size = 1 -> 2 bytes
    g1[1] = (int)((td0 & 0xFFFFu) << 16);           // tensor_dim0[15:0]
    g1[2] = (int)((td0 >> 16) | ((td1 & 0xFFFFu) << 16));
    g1[3] = (int)((td1 >> 16) | (tile_w << 16));    // tile_dim0
    g1[4] = (int)(tile_h & 0xFFFFu);                // tile_dim1 (tile_dim2 = 0)
    g1[5] = (int)(unsigned)(row_stride_elems & 0xFFFFFFFFu);   // dim0_stride[31:0]
    g1[6] = (int)(unsigned)((row_stride_elems >> 32) & 0xFFFFu);
    g1[7] = 0;
    i32x4 z = {0, 0, 0, 0};
#if defined(__clang_major__) && (__clang_major__ >= 23)
    i32x8 z8 = {0, 0, 0, 0, 0, 0, 0, 0};
    __builtin_amdgcn_tensor_load_to_lds(g0, g1, z, z, z8, 0);
#else
    __builtin_amdgcn_tensor_load_to_lds(g0, g1, z, z, 0);
#endif
#else
    int lane = threadIdx.x & 31;
    for (unsigned i = lane; i < tile_w * tile_h; i += 32)
        lds_dst[i] = gsrc[(size_t)(i / tile_w) * row_stride_elems + (i % tile_w)];
#endif
}

__device__ __forceinline__ void tdm_wait(int n) {
#ifdef HAVE_TDM
    if (n == 0) __builtin_amdgcn_s_wait_tensorcnt((short)0);
    else        __builtin_amdgcn_s_wait_tensorcnt((short)2);
#endif
    asm volatile("" ::: "memory");
}

// ---------- stage 1: conversions ----------
__global__ void mha_cvt_bf16(const float* __restrict__ in,
                             unsigned short* __restrict__ out, int n) {
    int i = blockIdx.x * 256 + threadIdx.x;
    if (i < n) out[i] = f2bf(in[i]);
}

// in: [H, E, D] f32  ->  out: [H, D, E] bf16
__global__ void mha_transpose_bf16(const float* __restrict__ in,
                                   unsigned short* __restrict__ out,
                                   int E, int D) {
    int h = blockIdx.y;
    int i = blockIdx.x * 256 + threadIdx.x;
    if (i >= E * D) return;
    int e = i / D, d = i % D;
    out[(size_t)h * D * E + (size_t)d * E + e] =
        f2bf(in[(size_t)h * E * D + (size_t)e * D + d]);
}

// ---------- stage 2: QKV projection (32x64 tile, ping-pong pipelined) ----------
__global__ void __launch_bounds__(32) mha_qkv(
    const unsigned short* __restrict__ xb,   // [B*S, E] bf16
    const unsigned short* __restrict__ wqt,  // [H, D, E] bf16
    const unsigned short* __restrict__ wkt,
    const unsigned short* __restrict__ wvt,
    unsigned short* __restrict__ q,          // [B,H,S,D] bf16
    unsigned short* __restrict__ k,          // [B,H,S,D] bf16
    unsigned short* __restrict__ vt)         // [B,H,D,S] bf16
{
    int lane = threadIdx.x;
    int it = blockIdx.x, h = blockIdx.y, z = blockIdx.z;
    int b = z / 3, which = z % 3;
    int s0 = it * 32;

    const unsigned short* x0 = xb + (size_t)b * SEQ_LEN * N_EMBED;
    const unsigned short* wt =
        (which == 0 ? wqt : (which == 1 ? wkt : wvt)) +
        (size_t)h * HEAD_SIZE * N_EMBED;

    v8f acc[8];
#pragma unroll
    for (int i = 0; i < 8; i++)
#pragma unroll
        for (int j = 0; j < 8; j++) acc[i][j] = 0.0f;

    // ping-pong register sets: no cross-iteration copies, loads overlap WMMAs
    v16bf aL0, aH0, b0[4], aL1, aH1, b1[4];
    aL0 = ldfrag(x0, N_EMBED, s0,      0, lane);
    aH0 = ldfrag(x0, N_EMBED, s0 + 16, 0, lane);
#pragma unroll
    for (int nt = 0; nt < 4; nt++) b0[nt] = ldfrag(wt, N_EMBED, nt * 16, 0, lane);

    for (int e0 = 0; e0 < N_EMBED; e0 += 64) {
        // prefetch stage 1 (e0+32 is always in range: last e0 = 704)
        aL1 = ldfrag(x0, N_EMBED, s0,      e0 + 32, lane);
        aH1 = ldfrag(x0, N_EMBED, s0 + 16, e0 + 32, lane);
#pragma unroll
        for (int nt = 0; nt < 4; nt++)
            b1[nt] = ldfrag(wt, N_EMBED, nt * 16, e0 + 32, lane);
        // compute stage 0
#pragma unroll
        for (int nt = 0; nt < 4; nt++) {
            acc[nt]     = wmma_bf16(aL0, b0[nt], acc[nt]);
            acc[4 + nt] = wmma_bf16(aH0, b0[nt], acc[4 + nt]);
        }
        // prefetch stage 0 of next double-step
        if (e0 + 64 < N_EMBED) {
            aL0 = ldfrag(x0, N_EMBED, s0,      e0 + 64, lane);
            aH0 = ldfrag(x0, N_EMBED, s0 + 16, e0 + 64, lane);
#pragma unroll
            for (int nt = 0; nt < 4; nt++)
                b0[nt] = ldfrag(wt, N_EMBED, nt * 16, e0 + 64, lane);
        }
        // compute stage 1
#pragma unroll
        for (int nt = 0; nt < 4; nt++) {
            acc[nt]     = wmma_bf16(aL1, b1[nt], acc[nt]);
            acc[4 + nt] = wmma_bf16(aH1, b1[nt], acc[4 + nt]);
        }
    }

    int n = lane & 15, half = lane >> 4;
    if (which < 2) {
        unsigned short* out =
            (which == 0 ? q : k) +
            (size_t)(b * NUM_HEADS + h) * SEQ_LEN * HEAD_SIZE;
#pragma unroll
        for (int g = 0; g < 2; g++)
#pragma unroll
            for (int nt = 0; nt < 4; nt++)
#pragma unroll
                for (int r = 0; r < 8; r++) {
                    int s = s0 + g * 16 + r + 8 * half;
                    out[(size_t)s * HEAD_SIZE + nt * 16 + n] = f2bf(acc[g * 4 + nt][r]);
                }
    } else {
        unsigned short* out =
            vt + (size_t)(b * NUM_HEADS + h) * HEAD_SIZE * SEQ_LEN;
#pragma unroll
        for (int g = 0; g < 2; g++)
#pragma unroll
            for (int nt = 0; nt < 4; nt++)
#pragma unroll
                for (int r = 0; r < 8; r++) {
                    int s = s0 + g * 16 + r + 8 * half;
                    out[(size_t)(nt * 16 + n) * SEQ_LEN + s] = f2bf(acc[g * 4 + nt][r]);
                }
    }
}

// ---------- stage 3: causal flash attention, TDM-staged K/V ----------
__global__ void __launch_bounds__(32) mha_attn(
    const unsigned short* __restrict__ q,   // [B,H,S,D]
    const unsigned short* __restrict__ k,   // [B,H,S,D]
    const unsigned short* __restrict__ vt,  // [B,H,D,S]
    unsigned short* __restrict__ ocat)      // [B,S,H*D] bf16
{
    __shared__ __align__(16) unsigned short kbuf[2][32 * 64]; // [t][d] chunk
    __shared__ __align__(16) unsigned short vbuf[2][64 * 32]; // [d][t] chunk
    __shared__ __align__(16) unsigned short pbuf[16 * 32];    // P transpose tile

    int lane = threadIdx.x;
    int it = blockIdx.x, h = blockIdx.y, b = blockIdx.z;
    int s0 = it * 16;
    int n = lane & 15, half = lane >> 4;

    const unsigned short* qb = q  + (size_t)(b * NUM_HEADS + h) * SEQ_LEN * HEAD_SIZE;
    const unsigned short* kb = k  + (size_t)(b * NUM_HEADS + h) * SEQ_LEN * HEAD_SIZE;
    const unsigned short* vb = vt + (size_t)(b * NUM_HEADS + h) * HEAD_SIZE * SEQ_LEN;

    v16bf aq0 = ldfrag(qb, HEAD_SIZE, s0, 0, lane);
    v16bf aq1 = ldfrag(qb, HEAD_SIZE, s0, 32, lane);

    v8f acc[4];
    float mr[8], lr[8];
#pragma unroll
    for (int i = 0; i < 4; i++)
#pragma unroll
        for (int j = 0; j < 8; j++) acc[i][j] = 0.0f;
#pragma unroll
    for (int r = 0; r < 8; r++) { mr[r] = -1e30f; lr[r] = 0.0f; }

    const float sc = 0.125f * LOG2E;  // 1/sqrt(64), log2 domain
    int nch = (s0 + 47) / 32;         // chunks of 32 keys, covering t <= s0+15

    // preload chunk 0
    tdm_load_tile(kbuf[0], kb, 64, 32, HEAD_SIZE);
    tdm_load_tile(vbuf[0], vb, 32, 64, SEQ_LEN);

    for (int c = 0; c < nch; c++) {
        int t0 = c * 32;
        if (c + 1 < nch) {
            // previous users of buffer (c+1)&1 were iteration c-1's ds reads
            asm volatile("s_wait_dscnt 0x0" ::: "memory");
            tdm_load_tile(kbuf[(c + 1) & 1], kb + (size_t)(t0 + 32) * HEAD_SIZE,
                          64, 32, HEAD_SIZE);
            tdm_load_tile(vbuf[(c + 1) & 1], vb + (t0 + 32),
                          32, 64, SEQ_LEN);
            tdm_wait(2);   // chunk c resident (in-order TDM completion)
        } else {
            tdm_wait(0);
        }
        const unsigned short* kt = kbuf[c & 1];
        const unsigned short* vl = vbuf[c & 1];

        // ---- scores: two 16x16 C-fragments over K = 64 (operands from LDS) ----
        v8f sf0, sf1;
#pragma unroll
        for (int j = 0; j < 8; j++) { sf0[j] = 0.0f; sf1[j] = 0.0f; }
        sf0 = wmma_bf16(aq0, ldfrag(kt, HEAD_SIZE, 0,  0,  lane), sf0);
        sf0 = wmma_bf16(aq1, ldfrag(kt, HEAD_SIZE, 0,  32, lane), sf0);
        sf1 = wmma_bf16(aq0, ldfrag(kt, HEAD_SIZE, 16, 0,  lane), sf1);
        sf1 = wmma_bf16(aq1, ldfrag(kt, HEAD_SIZE, 16, 32, lane), sf1);

        // ---- scale + causal mask ----
#pragma unroll
        for (int r = 0; r < 8; r++) {
            int srow = s0 + r + 8 * half;
            float v0 = sf0[r] * sc;
            float v1 = sf1[r] * sc;
            if (t0 + n      > srow) v0 = -1e30f;
            if (t0 + 16 + n > srow) v1 = -1e30f;
            sf0[r] = v0; sf1[r] = v1;
        }

        // ---- online softmax (row reductions across each 16-lane half) ----
        float corr[8];
#pragma unroll
        for (int r = 0; r < 8; r++) {
            float mx = fmaxf(sf0[r], sf1[r]);
            mx = fmaxf(mx, __shfl_xor(mx, 1, 32));
            mx = fmaxf(mx, __shfl_xor(mx, 2, 32));
            mx = fmaxf(mx, __shfl_xor(mx, 4, 32));
            mx = fmaxf(mx, __shfl_xor(mx, 8, 32));
            float mnew = fmaxf(mr[r], mx);
            float p0 = exp2f(sf0[r] - mnew);
            float p1 = exp2f(sf1[r] - mnew);
            sf0[r] = p0; sf1[r] = p1;
            float sum = p0 + p1;
            sum += __shfl_xor(sum, 1, 32);
            sum += __shfl_xor(sum, 2, 32);
            sum += __shfl_xor(sum, 4, 32);
            sum += __shfl_xor(sum, 8, 32);
            corr[r] = exp2f(mr[r] - mnew);
            lr[r] = lr[r] * corr[r] + sum;
            mr[r] = mnew;
        }
#pragma unroll
        for (int nt = 0; nt < 4; nt++)
#pragma unroll
            for (int r = 0; r < 8; r++) acc[nt][r] *= corr[r];

        // ---- transpose P (C-layout -> A-layout) through LDS ----
        __syncthreads();
#pragma unroll
        for (int r = 0; r < 8; r++) {
            pbuf[(r + 8 * half) * 32 + n]      = f2bf(sf0[r]);
            pbuf[(r + 8 * half) * 32 + 16 + n] = f2bf(sf1[r]);
        }
        __syncthreads();
        v16bf ap = ldfrag(pbuf, 32, 0, 0, lane);

        // ---- O += P * V (V operand from LDS) ----
#pragma unroll
        for (int nt = 0; nt < 4; nt++) {
            v16bf bv = ldfrag(vl, 32, nt * 16, 0, lane);
            acc[nt] = wmma_bf16(ap, bv, acc[nt]);
        }
    }

    // ---- normalize and store concat-head bf16 ----
#pragma unroll
    for (int r = 0; r < 8; r++) lr[r] = 1.0f / lr[r];
    unsigned short* ob = ocat + (size_t)b * SEQ_LEN * N_EMBED + h * HEAD_SIZE;
#pragma unroll
    for (int nt = 0; nt < 4; nt++)
#pragma unroll
        for (int r = 0; r < 8; r++) {
            int s = s0 + r + 8 * half;
            ob[(size_t)s * N_EMBED + nt * 16 + n] = f2bf(acc[nt][r] * lr[r]);
        }
}

// ---------- stage 4: output projection (32x64 tile, ping-pong pipelined) ----------
__global__ void __launch_bounds__(32) mha_proj(
    const unsigned short* __restrict__ ocat,  // [B*S, 768] bf16
    const unsigned short* __restrict__ wpt,   // [768(n), 768(k)] bf16
    const float* __restrict__ bp,
    float* __restrict__ out)                  // [B*S, 768] f32
{
    int lane = threadIdx.x;
    int r0 = blockIdx.x * 32;
    int c0 = blockIdx.y * 64;

    v8f acc[8];
#pragma unroll
    for (int i = 0; i < 8; i++)
#pragma unroll
        for (int j = 0; j < 8; j++) acc[i][j] = 0.0f;

    v16bf aL0, aH0, b0[4], aL1, aH1, b1[4];
    aL0 = ldfrag(ocat, N_EMBED, r0,      0, lane);
    aH0 = ldfrag(ocat, N_EMBED, r0 + 16, 0, lane);
#pragma unroll
    for (int nt = 0; nt < 4; nt++) b0[nt] = ldfrag(wpt, N_EMBED, c0 + nt * 16, 0, lane);

    for (int k0 = 0; k0 < N_EMBED; k0 += 64) {
        aL1 = ldfrag(ocat, N_EMBED, r0,      k0 + 32, lane);
        aH1 = ldfrag(ocat, N_EMBED, r0 + 16, k0 + 32, lane);
#pragma unroll
        for (int nt = 0; nt < 4; nt++)
            b1[nt] = ldfrag(wpt, N_EMBED, c0 + nt * 16, k0 + 32, lane);
#pragma unroll
        for (int nt = 0; nt < 4; nt++) {
            acc[nt]     = wmma_bf16(aL0, b0[nt], acc[nt]);
            acc[4 + nt] = wmma_bf16(aH0, b0[nt], acc[4 + nt]);
        }
        if (k0 + 64 < N_EMBED) {
            aL0 = ldfrag(ocat, N_EMBED, r0,      k0 + 64, lane);
            aH0 = ldfrag(ocat, N_EMBED, r0 + 16, k0 + 64, lane);
#pragma unroll
            for (int nt = 0; nt < 4; nt++)
                b0[nt] = ldfrag(wpt, N_EMBED, c0 + nt * 16, k0 + 64, lane);
        }
#pragma unroll
        for (int nt = 0; nt < 4; nt++) {
            acc[nt]     = wmma_bf16(aL1, b1[nt], acc[nt]);
            acc[4 + nt] = wmma_bf16(aH1, b1[nt], acc[4 + nt]);
        }
    }

    int n = lane & 15, half = lane >> 4;
#pragma unroll
    for (int g = 0; g < 2; g++)
#pragma unroll
        for (int nt = 0; nt < 4; nt++)
#pragma unroll
            for (int r = 0; r < 8; r++) {
                int col = c0 + nt * 16 + n;
                out[(size_t)(r0 + g * 16 + r + 8 * half) * N_EMBED + col] =
                    acc[g * 4 + nt][r] + bp[col];
            }
}

// ---------- launch ----------
extern "C" void kernel_launch(void* const* d_in, const int* in_sizes, int n_in,
                              void* d_out, int out_size, void* d_ws, size_t ws_size,
                              hipStream_t stream) {
    const float* x  = (const float*)d_in[0];
    const float* Wq = (const float*)d_in[1];
    const float* Wk = (const float*)d_in[2];
    const float* Wv = (const float*)d_in[3];
    const float* Wp = (const float*)d_in[4];
    const float* bp = (const float*)d_in[5];
    float* out = (float*)d_out;

    const size_t XN = (size_t)BATCH * SEQ_LEN * N_EMBED;
    const size_t WN = (size_t)NUM_HEADS * N_EMBED * HEAD_SIZE;
    const size_t PN = (size_t)N_EMBED * N_EMBED;
    const size_t QN = (size_t)BATCH * NUM_HEADS * SEQ_LEN * HEAD_SIZE;

    char* ws = (char*)d_ws;
    size_t off = 0;
    unsigned short* xb  = (unsigned short*)(ws + off); off += XN * 2;  // reused as ocat
    unsigned short* wqt = (unsigned short*)(ws + off); off += WN * 2;
    unsigned short* wkt = (unsigned short*)(ws + off); off += WN * 2;
    unsigned short* wvt = (unsigned short*)(ws + off); off += WN * 2;
    unsigned short* wpt = (unsigned short*)(ws + off); off += PN * 2;
    unsigned short* qb  = (unsigned short*)(ws + off); off += QN * 2;
    unsigned short* kb  = (unsigned short*)(ws + off); off += QN * 2;
    unsigned short* vtb = (unsigned short*)(ws + off); off += QN * 2;
    unsigned short* ocat = xb;   // xb dead after mha_qkv

    mha_cvt_bf16<<<(XN + 255) / 256, 256, 0, stream>>>(x, xb, (int)XN);
    {
        dim3 g((N_EMBED * HEAD_SIZE + 255) / 256, NUM_HEADS);
        mha_transpose_bf16<<<g, 256, 0, stream>>>(Wq, wqt, N_EMBED, HEAD_SIZE);
        mha_transpose_bf16<<<g, 256, 0, stream>>>(Wk, wkt, N_EMBED, HEAD_SIZE);
        mha_transpose_bf16<<<g, 256, 0, stream>>>(Wv, wvt, N_EMBED, HEAD_SIZE);
    }
    {
        dim3 g((N_EMBED * N_EMBED + 255) / 256, 1);
        mha_transpose_bf16<<<g, 256, 0, stream>>>(Wp, wpt, N_EMBED, N_EMBED);
    }
    {
        dim3 g(SEQ_LEN / 32, NUM_HEADS, BATCH * 3);
        mha_qkv<<<g, 32, 0, stream>>>(xb, wqt, wkt, wvt, qb, kb, vtb);
    }
    {
        dim3 g(SEQ_LEN / 16, NUM_HEADS, BATCH);
        mha_attn<<<g, 32, 0, stream>>>(qb, kb, vtb, ocat);
    }
    {
        dim3 g((BATCH * SEQ_LEN) / 32, N_EMBED / 64);
        mha_proj<<<g, 32, 0, stream>>>(ocat, wpt, bp, out);
    }
}
